// DeformableTransformerEncoderLayer_19791209299999
// MI455X (gfx1250) — matrix operand order repaired and checked by reading, not compiled, not run on hardware
//
#include <hip/hip_runtime.h>
#include <hip/hip_bf16.h>
#include <math.h>

// ---------------- problem constants ----------------
#define D_MODEL   256
#define D_FFN     1024
#define N_HEADS   8
#define N_LEVELS  4
#define N_POINTS  4
#define HEAD_DIM  32
#define BATCH     2
#define SEQ_LEN   13294
#define M_ROWS    (BATCH * SEQ_LEN)   // 26588
#define M_PAD     26624               // multiple of 128

typedef __attribute__((ext_vector_type(16))) __bf16 v16bf;
typedef __attribute__((ext_vector_type(8)))  __bf16 v8bf;
typedef __attribute__((ext_vector_type(8)))  float  v8f;

// ---------------------------------------------------------------------------
// prep: q = (src + pos) -> bf16 ; src -> bf16 ; zero pad rows
// grid = M_PAD blocks of 256 threads (one row per block)
// ---------------------------------------------------------------------------
__global__ void prep_qsrc(const float* __restrict__ src, const float* __restrict__ pos,
                          __bf16* __restrict__ qb, __bf16* __restrict__ sb) {
    int row = blockIdx.x;
    int c   = threadIdx.x;
    size_t idx = (size_t)row * D_MODEL + c;
    float s = 0.f, p = 0.f;
    if (row < M_ROWS) { s = src[idx]; p = pos[idx]; }
    qb[idx] = (__bf16)(s + p);
    sb[idx] = (__bf16)s;
}

// ---------------------------------------------------------------------------
// f32 -> bf16 convert (weights)
// ---------------------------------------------------------------------------
__global__ void f32_to_bf16(const float* __restrict__ in, __bf16* __restrict__ out, int n) {
    int i = blockIdx.x * blockDim.x + threadIdx.x;
    if (i < n) out[i] = (__bf16)in[i];
}

// ---------------------------------------------------------------------------
// WMMA bf16 GEMM:  C[M,N] = A[M,K] (row-major bf16) x W[N,K]^T (row-major bf16) + bias
// Block: 128 threads = 4 waves. Block tile 128(M) x 64(N); wave tile 32(M) x 64(N)
// = 8 WMMAs per k-step per wave. K stepped by 32.
// Both A and B tiles staged once per block through LDS (A rows copied
// row-major with no swizzle, B rows are the B-fragment lane map directly since the
// NT layout keeps K contiguous). Next k-step's tiles are prefetched with
// global_prefetch_b8 while the current WMMAs run.
// MODE 0: store f32 (C + bias).   MODE 1: relu(C + bias) stored as bf16.
// Requires: M % 128 == 0, N % 64 == 0, K % 32 == 0.
// ---------------------------------------------------------------------------
template<int MODE>
__global__ void gemm_bf16_wmma(const __bf16* __restrict__ A, const __bf16* __restrict__ W,
                               const float* __restrict__ bias, void* __restrict__ Cout,
                               int M, int N, int K) {
    __shared__ __align__(32) __bf16 As[128 * 32];   // 8 KB
    __shared__ __align__(32) __bf16 Bs[64 * 32];    // 4 KB

    const int tid  = threadIdx.x;
    const int wave = tid >> 5;
    const int lane = tid & 31;
    const int hi   = (lane >= 16) ? 1 : 0;
    const int l15  = lane & 15;
    const int m0   = blockIdx.x * 128;
    const int n0   = blockIdx.y * 64;

    v8f acc[2][4];
#pragma unroll
    for (int mt = 0; mt < 2; ++mt)
#pragma unroll
        for (int t = 0; t < 4; ++t)
#pragma unroll
            for (int e = 0; e < 8; ++e) acc[mt][t][e] = 0.f;

    // cooperative load maps:
    //   A tile 128x32 bf16 = 8KB: thread tid loads row tid, 32 elems (2x v16bf)
    //   B tile  64x32 bf16 = 4KB: thread tid loads row tid>>1, half (tid&1)*16
    const int b_row = tid >> 1;
    const int b_col = (tid & 1) * 16;

    const int nk = K >> 5;
    for (int kt = 0; kt < nk; ++kt) {
        const __bf16* ag = A + (size_t)(m0 + tid) * K + kt * 32;
        const __bf16* bg = W + (size_t)(n0 + b_row) * K + kt * 32 + b_col;

        __syncthreads();  // previous iteration's LDS reads complete
        *(v16bf*)(&As[tid * 32])          = *(const v16bf*)(ag);
        *(v16bf*)(&As[tid * 32 + 16])     = *(const v16bf*)(ag + 16);
        *(v16bf*)(&Bs[b_row * 32 + b_col]) = *(const v16bf*)(bg);
        if (kt + 1 < nk) {                 // pull next tiles toward the WGP
            __builtin_prefetch(ag + 32, 0, 3);
            __builtin_prefetch(bg + 32, 0, 3);
        }
        __syncthreads();

        // A fragments (contiguous k order, same as B below):
        // lane<16: K = 0..15; lane>=16: K = 16..31
        const int koff = hi ? 16 : 0;
        union { v16bf v; v8bf h[2]; } af[2];
#pragma unroll
        for (int mt = 0; mt < 2; ++mt) {
            const int arow = (wave * 32 + mt * 16 + l15) * 32;
            af[mt].h[0] = *(const v8bf*)(&As[arow + koff]);
            af[mt].h[1] = *(const v8bf*)(&As[arow + koff + 8]);
        }

        // B fragments: lane<16 -> N=l15, K=0..15 ; lane>=16 -> K=16..31
        const int kb = hi ? 16 : 0;
#pragma unroll
        for (int t = 0; t < 4; ++t) {
            v16bf bf = *(const v16bf*)(&Bs[(t * 16 + l15) * 32 + kb]);
#pragma unroll
            for (int mt = 0; mt < 2; ++mt) {
                acc[mt][t] = __builtin_amdgcn_wmma_f32_16x16x32_bf16(
                    false, af[mt].v, false, bf, (short)0, acc[mt][t], false, false);
            }
        }
    }

    // epilogue: C/D layout -> lane<16: N=l15, M=v ; lane>=16: N=l15, M=v+8
#pragma unroll
    for (int t = 0; t < 4; ++t) {
        const int n = n0 + t * 16 + l15;
        const float bv = bias[n];
#pragma unroll
        for (int mt = 0; mt < 2; ++mt) {
#pragma unroll
            for (int v = 0; v < 8; ++v) {
                const int m = m0 + wave * 32 + mt * 16 + v + (hi ? 8 : 0);
                float val = acc[mt][t][v] + bv;
                if (MODE == 1) {
                    val = val > 0.f ? val : 0.f;
                    ((__bf16*)Cout)[(size_t)m * N + n] = (__bf16)val;
                } else {
                    ((float*)Cout)[(size_t)m * N + n] = val;
                }
            }
        }
    }
}

// ---------------------------------------------------------------------------
// softmax over 16 (levels*points) per (row, head). One thread per (row, head).
// ---------------------------------------------------------------------------
__global__ void softmax16(const float* __restrict__ in, float* __restrict__ out, int nrh) {
    int i = blockIdx.x * blockDim.x + threadIdx.x;
    if (i >= nrh) return;
    const float* p = in + (size_t)i * 16;
    float mx = p[0];
#pragma unroll
    for (int k = 1; k < 16; ++k) mx = fmaxf(mx, p[k]);
    float e[16], s = 0.f;
#pragma unroll
    for (int k = 0; k < 16; ++k) { e[k] = __expf(p[k] - mx); s += e[k]; }
    float inv = 1.f / s;
    float* o = out + (size_t)i * 16;
#pragma unroll
    for (int k = 0; k < 16; ++k) o[k] = e[k] * inv;
}

// ---------------------------------------------------------------------------
// deformable attention sampling core.
// One wave per query; lane = head-dim channel (HEAD_DIM == 32 == wave32).
// value (27 MB) is fully L2-resident, so the scattered 128B/wave gathers are
// L2-bandwidth traffic. Output written as bf16 (feeds the output-proj GEMM).
// ---------------------------------------------------------------------------
__global__ void deform_sample(const float* __restrict__ value,     // [M_PAD,256]
                              const float* __restrict__ offsets,   // [M_PAD,256]
                              const float* __restrict__ attn,      // [M_PAD,128]
                              const float* __restrict__ refp,      // [B,LEN,4,2]
                              const long long* __restrict__ shapes,// [4,2]
                              const long long* __restrict__ lstart,// [4]
                              __bf16* __restrict__ sampled) {      // [M_PAD,256]
    const int wave = threadIdx.x >> 5;
    const int d    = threadIdx.x & 31;
    const int r    = blockIdx.x * 8 + wave;
    if (r >= M_PAD) return;

    if (r >= M_ROWS) {
#pragma unroll
        for (int h = 0; h < N_HEADS; ++h)
            sampled[(size_t)r * D_MODEL + h * HEAD_DIM + d] = (__bf16)0.f;
        return;
    }

    const int b = r / SEQ_LEN;
    const size_t orow = (size_t)r * D_MODEL;
    const size_t arow = (size_t)r * (N_HEADS * 16);
    const size_t rrow = (size_t)r * (N_LEVELS * 2);

    for (int h = 0; h < N_HEADS; ++h) {
        float acc = 0.f;
#pragma unroll
        for (int lvl = 0; lvl < N_LEVELS; ++lvl) {
            const int H   = (int)shapes[lvl * 2 + 0];
            const int W   = (int)shapes[lvl * 2 + 1];
            const int lsi = (int)lstart[lvl];
            const float rx = refp[rrow + lvl * 2 + 0];
            const float ry = refp[rrow + lvl * 2 + 1];
            const float fW = (float)W, fH = (float)H;
#pragma unroll
            for (int p = 0; p < N_POINTS; ++p) {
                const size_t oidx = orow + (size_t)(((h * N_LEVELS + lvl) * N_POINTS + p) * 2);
                const float ox = offsets[oidx + 0];
                const float oy = offsets[oidx + 1];
                const float x = (rx + ox / fW) * fW - 0.5f;
                const float y = (ry + oy / fH) * fH - 0.5f;
                const float x0f = floorf(x), y0f = floorf(y);
                const float wx = x - x0f, wy = y - y0f;
                const int x0 = (int)x0f, y0 = (int)y0f;
                const float a = attn[arow + h * 16 + lvl * N_POINTS + p];

                float g[4];
#pragma unroll
                for (int c = 0; c < 4; ++c) {
                    const int xi = x0 + (c & 1);
                    const int yi = y0 + (c >> 1);
                    float v = 0.f;
                    if (xi >= 0 && xi < W && yi >= 0 && yi < H) {
                        const size_t vrow = (size_t)(b * SEQ_LEN + lsi + yi * W + xi);
                        v = value[vrow * D_MODEL + h * HEAD_DIM + d];
                    }
                    g[c] = v;
                }
                const float top = g[0] * (1.f - wx) + g[1] * wx;
                const float bot = g[2] * (1.f - wx) + g[3] * wx;
                acc += a * (top * (1.f - wy) + bot * wy);
            }
        }
        sampled[orow + h * HEAD_DIM + d] = (__bf16)acc;
    }
}

// ---------------------------------------------------------------------------
// residual add + LayerNorm over 256 channels. One block (256 threads) per row.
// WRITE_BF16: also emit bf16 copy (feeds the FFN WMMA GEMM).
// ---------------------------------------------------------------------------
template<int WRITE_BF16>
__global__ void residual_ln(const float* __restrict__ a, const float* __restrict__ bsrc,
                            const float* __restrict__ g, const float* __restrict__ beta,
                            float* __restrict__ outf, __bf16* __restrict__ outb,
                            int validM) {
    const int row = blockIdx.x;
    const int t   = threadIdx.x;
    const size_t idx = (size_t)row * D_MODEL + t;

    float xa = (row < validM) ? a[idx] : 0.f;
    float x  = xa + bsrc[idx];

    __shared__ float s1[D_MODEL];
    __shared__ float s2[D_MODEL];
    s1[t] = x; s2[t] = x * x;
    __syncthreads();
    for (int st = 128; st > 0; st >>= 1) {
        if (t < st) { s1[t] += s1[t + st]; s2[t] += s2[t + st]; }
        __syncthreads();
    }
    const float mu  = s1[0] * (1.f / D_MODEL);
    const float var = s2[0] * (1.f / D_MODEL) - mu * mu;
    const float inv = rsqrtf(var + 1e-5f);
    const float y = (x - mu) * inv * g[t] + beta[t];
    if (outf) outf[idx] = y;
    if (WRITE_BF16) outb[idx] = (__bf16)y;
}

// ---------------------------------------------------------------------------
// launch
// ---------------------------------------------------------------------------
extern "C" void kernel_launch(void* const* d_in, const int* in_sizes, int n_in,
                              void* d_out, int out_size, void* d_ws, size_t ws_size,
                              hipStream_t stream) {
    const float*     src    = (const float*)d_in[0];
    const float*     pos    = (const float*)d_in[1];
    const float*     refp   = (const float*)d_in[2];
    const long long* shapes = (const long long*)d_in[3];
    const long long* lstart = (const long long*)d_in[4];
    const float* so_w = (const float*)d_in[5];
    const float* so_b = (const float*)d_in[6];
    const float* aw_w = (const float*)d_in[7];
    const float* aw_b = (const float*)d_in[8];
    const float* vp_w = (const float*)d_in[9];
    const float* vp_b = (const float*)d_in[10];
    const float* op_w = (const float*)d_in[11];
    const float* op_b = (const float*)d_in[12];
    const float* ln1_g = (const float*)d_in[13];
    const float* ln1_b = (const float*)d_in[14];
    const float* l1_w  = (const float*)d_in[15];
    const float* l1_b  = (const float*)d_in[16];
    const float* l2_w  = (const float*)d_in[17];
    const float* l2_b  = (const float*)d_in[18];
    const float* ln2_g = (const float*)d_in[19];
    const float* ln2_b = (const float*)d_in[20];

    // ---- workspace carve-up (256B aligned) ----
    char* wp = (char*)d_ws;
    auto carve = [&](size_t bytes) -> void* {
        void* r = (void*)wp;
        wp += (bytes + 255) & ~(size_t)255;
        return r;
    };
    __bf16* q_b    = (__bf16*)carve((size_t)M_PAD * D_MODEL * 2);
    __bf16* s_b    = (__bf16*)carve((size_t)M_PAD * D_MODEL * 2);
    __bf16* so_wb  = (__bf16*)carve((size_t)256 * 256 * 2);
    __bf16* aw_wb  = (__bf16*)carve((size_t)128 * 256 * 2);
    __bf16* vp_wb  = (__bf16*)carve((size_t)256 * 256 * 2);
    __bf16* op_wb  = (__bf16*)carve((size_t)256 * 256 * 2);
    __bf16* l1_wb  = (__bf16*)carve((size_t)D_FFN * 256 * 2);
    __bf16* l2_wb  = (__bf16*)carve((size_t)256 * D_FFN * 2);
    float*  value  = (float*)carve((size_t)M_PAD * D_MODEL * 4);
    float*  offs   = (float*)carve((size_t)M_PAD * D_MODEL * 4);
    float*  attn_r = (float*)carve((size_t)M_PAD * 128 * 4);
    float*  attn_s = (float*)carve((size_t)M_PAD * 128 * 4);
    __bf16* samp_b = (__bf16*)carve((size_t)M_PAD * D_MODEL * 2);
    float*  src2   = (float*)carve((size_t)M_PAD * D_MODEL * 4);
    float*  x1f    = (float*)carve((size_t)M_PAD * D_MODEL * 4);
    __bf16* x1b    = (__bf16*)carve((size_t)M_PAD * D_MODEL * 2);
    __bf16* hid_b  = (__bf16*)carve((size_t)M_PAD * D_FFN * 2);
    float*  ffn    = (float*)carve((size_t)M_PAD * D_MODEL * 4);

    // 1) prep inputs (bf16, zero padded rows)
    prep_qsrc<<<M_PAD, 256, 0, stream>>>(src, pos, q_b, s_b);

    // 2) weight conversion
    auto conv = [&](const float* in, __bf16* out, int n) {
        f32_to_bf16<<<(n + 255) / 256, 256, 0, stream>>>(in, out, n);
    };
    conv(so_w, so_wb, 256 * 256);
    conv(aw_w, aw_wb, 128 * 256);
    conv(vp_w, vp_wb, 256 * 256);
    conv(op_w, op_wb, 256 * 256);
    conv(l1_w, l1_wb, D_FFN * 256);
    conv(l2_w, l2_wb, 256 * D_FFN);

    // 3) input projections (WMMA bf16)
    gemm_bf16_wmma<0><<<dim3(M_PAD / 128, 256 / 64), 128, 0, stream>>>(
        s_b, vp_wb, vp_b, (void*)value, M_PAD, 256, 256);
    gemm_bf16_wmma<0><<<dim3(M_PAD / 128, 256 / 64), 128, 0, stream>>>(
        q_b, so_wb, so_b, (void*)offs, M_PAD, 256, 256);
    gemm_bf16_wmma<0><<<dim3(M_PAD / 128, 128 / 64), 128, 0, stream>>>(
        q_b, aw_wb, aw_b, (void*)attn_r, M_PAD, 128, 256);

    // 4) softmax over (levels*points) per head
    softmax16<<<(M_PAD * N_HEADS + 255) / 256, 256, 0, stream>>>(
        attn_r, attn_s, M_PAD * N_HEADS);

    // 5) deformable sampling (one wave per query; lane = channel)
    deform_sample<<<M_PAD / 8, 256, 0, stream>>>(
        value, offs, attn_s, refp, shapes, lstart, samp_b);

    // 6) output projection
    gemm_bf16_wmma<0><<<dim3(M_PAD / 128, 256 / 64), 128, 0, stream>>>(
        samp_b, op_wb, op_b, (void*)src2, M_PAD, 256, 256);

    // 7) residual + LN1 (emit f32 + bf16)
    residual_ln<1><<<M_PAD, 256, 0, stream>>>(src, src2, ln1_g, ln1_b, x1f, x1b, M_ROWS);

    // 8) FFN: relu(x @ l1^T + b1) as bf16, then @ l2^T + b2 as f32
    gemm_bf16_wmma<1><<<dim3(M_PAD / 128, D_FFN / 64), 128, 0, stream>>>(
        x1b, l1_wb, l1_b, (void*)hid_b, M_PAD, D_FFN, 256);
    gemm_bf16_wmma<0><<<dim3(M_PAD / 128, 256 / 64), 128, 0, stream>>>(
        hid_b, l2_wb, l2_b, (void*)ffn, M_PAD, 256, D_FFN);

    // 9) residual + LN2 -> final output (only real rows)
    residual_ln<0><<<M_ROWS, 256, 0, stream>>>(x1f, ffn, ln2_g, ln2_b,
                                               (float*)d_out, (__bf16*)nullptr, M_PAD);
}
